// HarmonicOscillator_79731772883057
// MI455X (gfx1250) — matrix operand order repaired and checked by reading, not compiled, not run on hardware
//
#include <hip/hip_runtime.h>

typedef __attribute__((ext_vector_type(16))) _Float16 v16h;
typedef __attribute__((ext_vector_type(8)))  float    v8f;

#define SR        22050.0f
#define NH        64
#define HOP       512
#define NFRAMES   512
#define NBATCH    8
#define NSAMP     (NFRAMES * HOP)   // 262144 samples per batch
#define CHUNK     512
#define NCHUNK    (NSAMP / CHUNK)   // 512 chunks per batch
#define SCALEF    (511.0f / 262143.0f)  // (n_in-1)/(n_out-1), align_corners
#define TWO_PI    6.28318530717958647692f

__device__ __forceinline__ float fracf_(float x) { return x - floorf(x); }

// f0 upsample (align_corners linear interp), in cycles/sample units (divide by SR)
__device__ __forceinline__ float f0_cyc(const float* __restrict__ f0b, int n) {
    float pos = (float)n * SCALEF;
    int i0 = (int)floorf(pos);
    if (i0 > NFRAMES - 1) i0 = NFRAMES - 1;
    int i1 = min(i0 + 1, NFRAMES - 1);
    float w = pos - (float)i0;
    return (f0b[i0] * (1.0f - w) + f0b[i1] * w) * (1.0f / SR);
}

// ---------------------------------------------------------------------------
// Kernel 1: per-chunk sum of f0_up/SR (fractional part only — phase is mod 1)
// ---------------------------------------------------------------------------
__launch_bounds__(256)
__global__ void ho_chunk_sums(const float* __restrict__ f0, float* __restrict__ csum) {
    __shared__ float red[256];
    int blk = blockIdx.x;
    int b = blk >> 9;                 // NCHUNK == 512
    int c = blk & (NCHUNK - 1);
    const float* f0b = f0 + b * NFRAMES;
    int tid = threadIdx.x;
    int n0 = c * CHUNK;
    float v = f0_cyc(f0b, n0 + 2 * tid) + f0_cyc(f0b, n0 + 2 * tid + 1);
    red[tid] = v;
    __syncthreads();
    #pragma unroll
    for (int s = 128; s > 0; s >>= 1) {
        if (tid < s) red[tid] += red[tid + s];
        __syncthreads();
    }
    if (tid == 0) csum[blk] = fracf_(red[0]);
}

// ---------------------------------------------------------------------------
// Kernel 2: per-batch exclusive scan of chunk sums (mod 1). 4096 adds total.
// ---------------------------------------------------------------------------
__global__ void ho_chunk_scan(const float* __restrict__ csum, float* __restrict__ expref) {
    if (threadIdx.x != 0) return;
    int b = blockIdx.x;
    float run = 0.0f;
    for (int c = 0; c < NCHUNK; ++c) {
        expref[b * NCHUNK + c] = run;
        run = fracf_(run + csum[b * NCHUNK + c]);
    }
}

// ---------------------------------------------------------------------------
// Kernel 3: intra-chunk scan -> phase q (cycles, mod 1); Chebyshev sine
// recurrence builds S tiles (16 samples x 64 harmonics, f16); WMMA
// v_wmma_f32_16x16x32_f16 computes D = S x A^T against the <=4 frame rows the
// chunk touches; each sample lerps two columns of D for the final audio.
// ---------------------------------------------------------------------------
__launch_bounds__(256)
__global__ void ho_synth(const float* __restrict__ f0, const float* __restrict__ amps,
                         const float* __restrict__ expref, float* __restrict__ out) {
    __shared__ float sScan[256];
    __shared__ float sQ[CHUNK];
    __shared__ float sD[8 * 256];     // per-wave 16x16 f32 D staging

    int blk = blockIdx.x;
    int b = blk >> 9;
    int c = blk & (NCHUNK - 1);
    const float* f0b = f0 + b * NFRAMES;
    int tid  = threadIdx.x;
    int lane = tid & 31;
    int wave = tid >> 5;
    int n0 = c * CHUNK;
    float base = expref[blk];

    // --- intra-chunk inclusive scan of f0_up/SR (2 samples per thread) ---
    float d0 = f0_cyc(f0b, n0 + 2 * tid);
    float d1 = f0_cyc(f0b, n0 + 2 * tid + 1);
    float pair = d0 + d1;
    sScan[tid] = pair;
    __syncthreads();
    #pragma unroll
    for (int off = 1; off < 256; off <<= 1) {
        float v = sScan[tid];
        float add = (tid >= off) ? sScan[tid - off] : 0.0f;
        __syncthreads();
        sScan[tid] = v + add;
        __syncthreads();
    }
    float excl = sScan[tid] - pair;
    sQ[2 * tid]     = fracf_(base + excl + d0);
    sQ[2 * tid + 1] = fracf_(base + excl + d0 + d1);
    __syncthreads();

    // --- B fragments: columns = frames k0..k0+15 (only 0..3 ever selected) ---
    int k0  = (int)floorf((float)n0 * SCALEF);
    int col = lane & 15;
    int fr  = min(k0 + col, NFRAMES - 1);
    const float* arow = amps + (size_t)(b * NFRAMES + fr) * NH;
    int hbase = (lane < 16) ? 0 : 16;   // B layout: lanes 0-15 hold K=0..15, 16-31 hold K=16..31
    v16h bf0, bf1;
    #pragma unroll
    for (int j = 0; j < 16; ++j) {
        bf0[j] = (_Float16)arow[hbase + j];        // harmonics 1..32  (K = h-1)
        bf1[j] = (_Float16)arow[32 + hbase + j];   // harmonics 33..64
    }

    bool laneLo = (lane < 16);
    float* Dst = sD + wave * 256;
    int m = lane & 15;

    for (int t = 0; t < 4; ++t) {
        int tile = wave * 4 + t;
        int nl = tile * 16 + m;                  // lanes (m, m+16) share sample m
        float q  = sQ[nl];
        float th = TWO_PI * q;
        float s1 = __sinf(th);
        float c1 = __cosf(th);
        float twoC = 2.0f * c1;

        // Chebyshev recurrence: sin(h*th), h = 1..64 (compile-time indices -> regs)
        float s[NH + 1];
        s[1] = s1;
        s[2] = twoC * s1;
        #pragma unroll
        for (int h = 3; h <= NH; ++h) s[h] = twoC * s[h - 1] - s[h - 2];

        // Pack A fragments in contiguous K order (same as B):
        // lo lanes: K = 0..15; hi lanes: K = 16..31 per 32-K frag
        v16h a0, a1;
        #pragma unroll
        for (int j = 0; j < 16; ++j) {
            int kLo = j;
            int kHi = j + 16;
            a0[j] = (_Float16)(laneLo ? s[kLo + 1]  : s[kHi + 1]);
            a1[j] = (_Float16)(laneLo ? s[kLo + 33] : s[kHi + 33]);
        }

        v8f acc = {};
        acc = __builtin_amdgcn_wmma_f32_16x16x32_f16(false, a0, false, bf0,
                                                     (short)0, acc, false, false);
        acc = __builtin_amdgcn_wmma_f32_16x16x32_f16(false, a1, false, bf1,
                                                     (short)0, acc, false, false);

        // Stage D (16x16 f32) to this wave's LDS region
        int Mbase = laneLo ? 0 : 8;              // C/D layout: VGPR r -> row M = r (+8 for hi lanes)
        #pragma unroll
        for (int r = 0; r < 8; ++r) Dst[(Mbase + r) * 16 + (lane & 15)] = acc[r];

        // Per-sample lerp of two frame columns of D (in-wave LDS, in-order)
        if (laneLo) {
            int n = n0 + nl;
            float pos = (float)n * SCALEF;
            int i0 = (int)floorf(pos);
            if (i0 > NFRAMES - 1) i0 = NFRAMES - 1;
            int i1 = min(i0 + 1, NFRAMES - 1);
            float w = pos - (float)i0;
            int c0  = (i0 - k0) & 15;
            int cc1 = (i1 - k0) & 15;
            float audio = (1.0f - w) * Dst[m * 16 + c0] + w * Dst[m * 16 + cc1];
            out[(size_t)b * NSAMP + n] = audio;
        }
    }
}

// ---------------------------------------------------------------------------
extern "C" void kernel_launch(void* const* d_in, const int* in_sizes, int n_in,
                              void* d_out, int out_size, void* d_ws, size_t ws_size,
                              hipStream_t stream) {
    (void)in_sizes; (void)n_in; (void)out_size; (void)ws_size;
    const float* f0   = (const float*)d_in[0];   // (8, 512)
    const float* amps = (const float*)d_in[1];   // (8, 512, 64)
    float* out = (float*)d_out;                  // (8, 262144)

    float* csum   = (float*)d_ws;                // 4096 floats
    float* expref = csum + NBATCH * NCHUNK;      // 4096 floats

    ho_chunk_sums<<<NBATCH * NCHUNK, 256, 0, stream>>>(f0, csum);
    ho_chunk_scan<<<NBATCH, 32, 0, stream>>>(csum, expref);
    ho_synth<<<NBATCH * NCHUNK, 256, 0, stream>>>(f0, amps, expref, out);
}
